// PAM_Module_5609227289231
// MI455X (gfx1250) — matrix operand order repaired and checked by reading, not compiled, not run on hardware
//
#include <hip/hip_runtime.h>

typedef __attribute__((ext_vector_type(16))) __bf16 v16bf;
typedef __attribute__((ext_vector_type(8)))  float  v8f;

#define HW   4096
#define CIN  512
#define C8   64
#define NB   4

__device__ __forceinline__ v8f wmma_bf16(v16bf a, v16bf b, v8f c) {
    return __builtin_amdgcn_wmma_f32_16x16x32_bf16(false, a, false, b, (short)0, c, false, false);
}

// A-matrix fragment (16 x 32 bf16), row-major source, row stride ld elements.
// p points at element [m0=0][k0]. ISA layout: lane(m=lane&15, half=lane>>4)
// holds K = {8*half..8*half+7} in v[0..3] and K = {16+8*half..} in v[4..7].
__device__ __forceinline__ v16bf load_matA(const __bf16* p, int ld, int lane) {
    int m = lane & 15, h = lane >> 4;
    const __bf16* r = p + (size_t)m * ld + h * 8;
    union { v16bf v; uint4 q[2]; } u;
    u.q[0] = *(const uint4*)(r);
    u.q[1] = *(const uint4*)(r + 16);
    return u.v;
}

// B-matrix fragment (32 x 16 bf16), source stored column-major: column n is
// K-contiguous with column stride ldc elements. p points at [n0=0][k0].
// ISA layout: lane(n=lane&15, half=lane>>4) holds K = {8*half..8*half+7} in v[0..3] and K = {16+8*half..} in v[4..7].
__device__ __forceinline__ v16bf load_matB(const __bf16* p, int ldc, int lane) {
    int n = lane & 15, h = lane >> 4;
    const __bf16* r = p + (size_t)n * ldc + h * 8;
    union { v16bf v; uint4 q[2]; } u;
    u.q[0] = *(const uint4*)(r);
    u.q[1] = *(const uint4*)(r + 16);
    return u.v;
}

// -------------------------------------------------------------------------
// Y[b] = W (Mtot x 512) @ X[b] (512 x 4096) + bias.
// OUT_MODE 0: bf16 Y stored (B, 4096, Mtot)  [transposed: q/k]
// OUT_MODE 1: bf16 Y stored (B, Mtot, 4096)  [v]
// Tile: 64 (M) x 128 (N), 8 waves in 2x4, each wave 32x32 (2x2 WMMA tiles).
// -------------------------------------------------------------------------
template<int OUT_MODE>
__global__ __launch_bounds__(256)
void conv1x1_wmma(const float* __restrict__ W, const float* __restrict__ bias,
                  const float* __restrict__ X, __bf16* __restrict__ Y, int Mtot) {
    const int b  = blockIdx.z;
    const int n0 = blockIdx.x * 128;
    const int m0 = blockIdx.y * 64;
    const int tid = threadIdx.x, wave = tid >> 5, lane = tid & 31;
    const int wm = wave >> 2, wn = wave & 3;
    const int h = lane >> 4, nl = lane & 15;

    __shared__ __bf16 Asm[64 * 32];
    __shared__ __bf16 Bsm[128 * 32];

    v8f acc[2][2];
#pragma unroll
    for (int mi = 0; mi < 2; ++mi)
#pragma unroll
        for (int ni = 0; ni < 2; ++ni)
#pragma unroll
            for (int r = 0; r < 8; ++r) acc[mi][ni][r] = 0.f;

    const float* Xb = X + (size_t)b * CIN * HW;

    for (int k0 = 0; k0 < CIN; k0 += 32) {
        // stage A (64x32) fp32 -> bf16, row-major
        {
            int row = tid >> 2, kc = (tid & 3) * 8;
            const float* src = W + (size_t)(m0 + row) * CIN + k0 + kc;
            __bf16* dst = &Asm[row * 32 + kc];
#pragma unroll
            for (int j = 0; j < 8; ++j) dst[j] = (__bf16)src[j];
        }
        // stage B (32 k x 128 n) fp32 -> bf16, transposed to [n][k]
        {
            int kr = tid >> 3, nc = (tid & 7) * 16;
            const float* src = Xb + (size_t)(k0 + kr) * HW + n0 + nc;
#pragma unroll
            for (int j = 0; j < 16; ++j) Bsm[(nc + j) * 32 + kr] = (__bf16)src[j];
        }
        __syncthreads();

        v16bf a0 = load_matA(&Asm[(wm * 32     ) * 32], 32, lane);
        v16bf a1 = load_matA(&Asm[(wm * 32 + 16) * 32], 32, lane);
        v16bf b0 = load_matB(&Bsm[(wn * 32     ) * 32], 32, lane);
        v16bf b1 = load_matB(&Bsm[(wn * 32 + 16) * 32], 32, lane);
        acc[0][0] = wmma_bf16(a0, b0, acc[0][0]);
        acc[0][1] = wmma_bf16(a0, b1, acc[0][1]);
        acc[1][0] = wmma_bf16(a1, b0, acc[1][0]);
        acc[1][1] = wmma_bf16(a1, b1, acc[1][1]);
        __syncthreads();
    }

#pragma unroll
    for (int mi = 0; mi < 2; ++mi)
#pragma unroll
        for (int ni = 0; ni < 2; ++ni)
#pragma unroll
            for (int r = 0; r < 8; ++r) {
                int gm = m0 + wm * 32 + mi * 16 + r + 8 * h;
                int gn = n0 + wn * 32 + ni * 16 + nl;
                float val = acc[mi][ni][r] + bias[gm];
                if (OUT_MODE == 0)
                    Y[((size_t)b * HW + gn) * Mtot + gm] = (__bf16)val;
                else
                    Y[((size_t)b * Mtot + gm) * HW + gn] = (__bf16)val;
            }
}

// -------------------------------------------------------------------------
// Fused energy + online-softmax + attention-apply (flash style).
// Workgroup: 16 query rows; 8 waves split 512 V-channels (64 each).
// j loop in blocks of 128; never materializes the 4096x4096 energy.
// q,k: (B,4096,64) bf16; v: (B,512,4096) bf16; writes out2 (B,4096,512) bf16
// with out2 = gamma * attn_out + x.
// -------------------------------------------------------------------------
__global__ __launch_bounds__(256)
void pam_attention(const __bf16* __restrict__ Q, const __bf16* __restrict__ K,
                   const __bf16* __restrict__ V, const float* __restrict__ X,
                   const float* __restrict__ gamma, __bf16* __restrict__ O2) {
    const int b  = blockIdx.y;
    const int i0 = blockIdx.x * 16;
    const int tid = threadIdx.x, wave = tid >> 5, lane = tid & 31;
    const int h = lane >> 4, nl = lane & 15;
    const int row = tid >> 4, seg = tid & 15;

    __shared__ float  Es[16 * 128];
    __shared__ __bf16 Pb[16 * 128];
    __shared__ float  red[256];
    __shared__ float  Mrow[16], Lrow[16], Scl[16];

    // Q fragments (16 queries x K=64 channels), reused for every j block
    const __bf16* qb = Q + ((size_t)b * HW + i0) * C8;
    v16bf qa0 = load_matA(qb, C8, lane);
    v16bf qa1 = load_matA(qb + 32, C8, lane);

    v8f acc[4];
#pragma unroll
    for (int n = 0; n < 4; ++n)
#pragma unroll
        for (int r = 0; r < 8; ++r) acc[n][r] = 0.f;

    if (tid < 16) { Mrow[tid] = -3.0e38f; Lrow[tid] = 0.f; }
    __syncthreads();

    for (int j0 = 0; j0 < HW; j0 += 128) {
        // ---- energy tile: wave w computes 16x16 for keys j0 + w*16 ----
        const __bf16* kb = K + ((size_t)b * HW + j0 + wave * 16) * C8;
        v16bf kb0 = load_matB(kb, C8, lane);       // K = 0..31
        v16bf kb1 = load_matB(kb + 32, C8, lane);  // K = 32..63
        v8f e;
#pragma unroll
        for (int r = 0; r < 8; ++r) e[r] = 0.f;
        e = wmma_bf16(qa0, kb0, e);
        e = wmma_bf16(qa1, kb1, e);
#pragma unroll
        for (int r = 0; r < 8; ++r)
            Es[(r + 8 * h) * 128 + wave * 16 + nl] = e[r];
        __syncthreads();

        // ---- online softmax over this 128-wide block ----
        const float* er = &Es[row * 128 + seg * 8];
        float lm = -3.0e38f;
#pragma unroll
        for (int c = 0; c < 8; ++c) lm = fmaxf(lm, er[c]);
        red[tid] = lm;
        __syncthreads();
        if (seg == 0) {
            float bm = red[row * 16];
#pragma unroll
            for (int t = 1; t < 16; ++t) bm = fmaxf(bm, red[row * 16 + t]);
            float mold = Mrow[row];
            float mnew = fmaxf(mold, bm);
            Scl[row]  = __expf(mold - mnew);
            Mrow[row] = mnew;
        }
        __syncthreads();
        float mnew = Mrow[row];
        float ls = 0.f;
#pragma unroll
        for (int c = 0; c < 8; ++c) {
            float p = __expf(er[c] - mnew);
            Pb[row * 128 + seg * 8 + c] = (__bf16)p;
            ls += p;
        }
        red[tid] = ls;
        __syncthreads();
        if (seg == 0) {
            float s = 0.f;
#pragma unroll
            for (int t = 0; t < 16; ++t) s += red[row * 16 + t];
            Lrow[row] = Lrow[row] * Scl[row] + s;
        }
        // rescale accumulators by per-row factor (C-tile: M = r + 8*h)
#pragma unroll
        for (int r = 0; r < 8; ++r) {
            float sc = Scl[r + 8 * h];
#pragma unroll
            for (int n = 0; n < 4; ++n) acc[n][r] *= sc;
        }

        // ---- apply: acc(16 x 64ch) += P(16x128) x V^T(128 x 16) tiles ----
#pragma unroll
        for (int kk = 0; kk < 4; ++kk) {
            v16bf pa = load_matA(&Pb[kk * 32], 128, lane);
#pragma unroll
            for (int n = 0; n < 4; ++n) {
                const __bf16* vcol =
                    V + ((size_t)b * CIN + wave * 64 + n * 16) * HW + j0 + kk * 32;
                v16bf vb = load_matB(vcol, HW, lane);
                acc[n] = wmma_bf16(pa, vb, acc[n]);
            }
        }
        __syncthreads();  // protect Es/Pb/Scl before next block
    }

    // ---- epilogue: /L, gamma*out + x, store (B,4096,512) bf16 ----
    float g = gamma[0];
#pragma unroll
    for (int n = 0; n < 4; ++n)
#pragma unroll
        for (int r = 0; r < 8; ++r) {
            int i = i0 + r + 8 * h;
            int c = wave * 64 + n * 16 + nl;
            float val = acc[n][r] / Lrow[r + 8 * h];
            val = g * val + X[((size_t)b * CIN + c) * HW + i];
            O2[((size_t)b * HW + i) * CIN + c] = (__bf16)val;
        }
}

// -------------------------------------------------------------------------
// pam_out[b,o,i] = sum_c Wd[o,c] * out2[b,i,c] + bd[o]   (fp32 result)
// out2 is (B,4096,512) bf16: B-tiles load directly (column i, K=c contig).
// -------------------------------------------------------------------------
__global__ __launch_bounds__(256)
void final_conv_wmma(const float* __restrict__ Wd, const float* __restrict__ bd,
                     const __bf16* __restrict__ Y2, float* __restrict__ Out) {
    const int b  = blockIdx.z;
    const int n0 = blockIdx.x * 128;
    const int m0 = blockIdx.y * 64;
    const int tid = threadIdx.x, wave = tid >> 5, lane = tid & 31;
    const int wm = wave >> 2, wn = wave & 3;
    const int h = lane >> 4, nl = lane & 15;

    __shared__ __bf16 Asm[64 * 32];

    v8f acc[2][2];
#pragma unroll
    for (int mi = 0; mi < 2; ++mi)
#pragma unroll
        for (int ni = 0; ni < 2; ++ni)
#pragma unroll
            for (int r = 0; r < 8; ++r) acc[mi][ni][r] = 0.f;

    const __bf16* Yb = Y2 + (size_t)b * HW * CIN;

    for (int k0 = 0; k0 < CIN; k0 += 32) {
        {
            int rrow = tid >> 2, kc = (tid & 3) * 8;
            const float* src = Wd + (size_t)(m0 + rrow) * CIN + k0 + kc;
            __bf16* dst = &Asm[rrow * 32 + kc];
#pragma unroll
            for (int j = 0; j < 8; ++j) dst[j] = (__bf16)src[j];
        }
        __syncthreads();
        v16bf a0 = load_matA(&Asm[(wm * 32     ) * 32], 32, lane);
        v16bf a1 = load_matA(&Asm[(wm * 32 + 16) * 32], 32, lane);
        v16bf b0 = load_matB(Yb + (size_t)(n0 + wn * 32     ) * CIN + k0, CIN, lane);
        v16bf b1 = load_matB(Yb + (size_t)(n0 + wn * 32 + 16) * CIN + k0, CIN, lane);
        acc[0][0] = wmma_bf16(a0, b0, acc[0][0]);
        acc[0][1] = wmma_bf16(a0, b1, acc[0][1]);
        acc[1][0] = wmma_bf16(a1, b0, acc[1][0]);
        acc[1][1] = wmma_bf16(a1, b1, acc[1][1]);
        __syncthreads();
    }

#pragma unroll
    for (int mi = 0; mi < 2; ++mi)
#pragma unroll
        for (int ni = 0; ni < 2; ++ni)
#pragma unroll
            for (int r = 0; r < 8; ++r) {
                int gm = m0 + wm * 32 + mi * 16 + r + 8 * h;
                int gn = n0 + wn * 32 + ni * 16 + nl;
                Out[((size_t)b * CIN + gm) * HW + gn] = acc[mi][ni][r] + bd[gm];
            }
}

extern "C" void kernel_launch(void* const* d_in, const int* in_sizes, int n_in,
                              void* d_out, int out_size, void* d_ws, size_t ws_size,
                              hipStream_t stream) {
    const float* x     = (const float*)d_in[0];
    const float* Wq    = (const float*)d_in[1];
    const float* bq    = (const float*)d_in[2];
    const float* Wk    = (const float*)d_in[3];
    const float* bk    = (const float*)d_in[4];
    const float* Wv    = (const float*)d_in[5];
    const float* bv    = (const float*)d_in[6];
    const float* gamma = (const float*)d_in[7];
    const float* Wd    = (const float*)d_in[8];
    const float* bd    = (const float*)d_in[9];
    float* out = (float*)d_out;

    // workspace layout (bf16): q 2MB | k 2MB | v 16MB | out2 16MB  = 36MB
    char* ws = (char*)d_ws;
    __bf16* qb = (__bf16*)(ws);
    __bf16* kb = (__bf16*)(ws + (size_t)NB * HW * C8 * 2);
    __bf16* vb = (__bf16*)(ws + (size_t)2 * NB * HW * C8 * 2);
    __bf16* o2 = (__bf16*)(ws + (size_t)2 * NB * HW * C8 * 2 + (size_t)NB * CIN * HW * 2);

    dim3 blk(256);
    conv1x1_wmma<0><<<dim3(HW / 128, 1,        NB), blk, 0, stream>>>(Wq, bq, x, qb, C8);
    conv1x1_wmma<0><<<dim3(HW / 128, 1,        NB), blk, 0, stream>>>(Wk, bk, x, kb, C8);
    conv1x1_wmma<1><<<dim3(HW / 128, CIN / 64, NB), blk, 0, stream>>>(Wv, bv, x, vb, CIN);
    pam_attention  <<<dim3(HW / 16, NB),            blk, 0, stream>>>(qb, kb, vb, x, gamma, o2);
    final_conv_wmma<<<dim3(HW / 128, CIN / 64, NB), blk, 0, stream>>>(Wd, bd, o2, out);
}
